// Model_42185168781873
// MI455X (gfx1250) — matrix operand order repaired and checked by reading, not compiled, not run on hardware
//
#include <hip/hip_runtime.h>

typedef __attribute__((ext_vector_type(16))) __bf16 v16bf;
typedef __attribute__((ext_vector_type(8)))  float  v8f;

#define N_NODES 10000
#define N_EDGES 160000
#define D       512          // IN_FEATS == H_FEATS
#define KTOT    1024         // 2*D
#define NCLS    128
#define MTILES  (N_NODES / 16)   // 625 (exact)
#define KTILES  (KTOT / 32)      // 32

// ---------------- utility: zero a 32-bit buffer (grid-stride) ----------------
__global__ void zero_b32(unsigned* __restrict__ p, int n) {
    int i = blockIdx.x * blockDim.x + threadIdx.x;
    int stride = gridDim.x * blockDim.x;
    for (; i < n; i += stride) p[i] = 0u;
}

// ---------------- in-degree histogram (int) ----------------
__global__ void degree_kernel(const int* __restrict__ dst, int* __restrict__ cnt) {
    int e = blockIdx.x * blockDim.x + threadIdx.x;
    if (e < N_EDGES) atomicAdd(&cnt[dst[e]], 1);
}

// ---------------- exclusive prefix scan of cnt -> row_start, cursor ----------
// single block, 1024 threads, 10 nodes/thread, LDS Hillis-Steele over partials
__global__ __launch_bounds__(1024)
void scan_kernel(const int* __restrict__ cnt, int* __restrict__ row_start,
                 int* __restrict__ cursor) {
    __shared__ int part[1024];
    const int t = threadIdx.x;
    const int PER = (N_NODES + 1023) / 1024;   // 10
    const int base = t * PER;
    int local[PER];
    int s = 0;
#pragma unroll
    for (int i = 0; i < PER; ++i) {
        int v = (base + i < N_NODES) ? cnt[base + i] : 0;
        local[i] = s;           // exclusive within chunk
        s += v;
    }
    part[t] = s;
    __syncthreads();
    for (int off = 1; off < 1024; off <<= 1) {
        int v = (t >= off) ? part[t - off] : 0;
        __syncthreads();
        part[t] += v;
        __syncthreads();
    }
    const int chunk_base = (t == 0) ? 0 : part[t - 1];
#pragma unroll
    for (int i = 0; i < PER; ++i) {
        int n = base + i;
        if (n < N_NODES) {
            int rs = chunk_base + local[i];
            row_start[n] = rs;
            cursor[n]    = rs;
        }
    }
}

// ---------------- bucket edges by dst (CSR build) ----------------
__global__ void bucket_kernel(const int* __restrict__ src, const int* __restrict__ dst,
                              int* __restrict__ cursor, int* __restrict__ edge_src) {
    int e = blockIdx.x * blockDim.x + threadIdx.x;
    if (e < N_EDGES) {
        int pos = atomicAdd(&cursor[dst[e]], 1);
        edge_src[pos] = src[e];
    }
}

// ---------------- per-node gather + mean: hN[n] = mean_{e: dst=n} x[src_e] ---
// one block (256 threads) per node; 2 features per thread; pure coalesced loads
__global__ __launch_bounds__(256)
void gather_mean(const float* __restrict__ x, const int* __restrict__ edge_src,
                 const int* __restrict__ row_start, const int* __restrict__ cnt,
                 float* __restrict__ hN) {
    const int n   = blockIdx.x;
    const int beg = row_start[n];
    const int deg = cnt[n];
    const int f0 = threadIdx.x;
    const int f1 = threadIdx.x + 256;
    float a0 = 0.0f, a1 = 0.0f;
    for (int i = 0; i < deg; ++i) {
        const float* row = x + (size_t)edge_src[beg + i] * D;
        a0 += row[f0];
        a1 += row[f1];
    }
    const float rinv = (deg > 0) ? (1.0f / (float)deg) : 0.0f;
    hN[(size_t)n * D + f0] = a0 * rinv;
    hN[(size_t)n * D + f1] = a1 * rinv;
}

// ---------------- pack weights into CDNA5 B-matrix (32x16 bf16) fragments ----
// lane L holds column n = L%16; lanes 0-15 hold K=0..15 (elem e -> K=e),
// lanes 16-31 hold K=16..31.  Index: ((nt*KTILES + kt)*32 + lane)*16.
__global__ void pack_B(const float* __restrict__ W, __bf16* __restrict__ Bp, int Ncols) {
    int t = blockIdx.x * blockDim.x + threadIdx.x;
    int total = (Ncols / 16) * KTILES * 32;
    if (t >= total) return;
    int lane = t & 31;
    int kt   = (t >> 5) % KTILES;
    int nt   = (t >> 5) / KTILES;
    int n     = nt * 16 + (lane & 15);
    int kbase = kt * 32 + ((lane < 16) ? 0 : 16);
    v16bf frag;
#pragma unroll
    for (int e = 0; e < 16; ++e)
        frag[e] = (__bf16)W[(size_t)(kbase + e) * Ncols + n];
    *reinterpret_cast<v16bf*>(Bp + (size_t)t * 16) = frag;
}

// ---------------- pack A = concat([h, hN]) into 16x32 bf16 A fragments ------
// lane L holds row m = L%16; lanes 0-15: elems 0-15 -> K 0-15;
// lanes 16-31: elems 0-15 -> K 16-31.
__global__ void pack_A(const float* __restrict__ h, const float* __restrict__ hN,
                       __bf16* __restrict__ Ap) {
    int t = blockIdx.x * blockDim.x + threadIdx.x;
    int total = MTILES * KTILES * 32;
    if (t >= total) return;
    int lane = t & 31;
    int kt   = (t >> 5) % KTILES;
    int mt   = (t >> 5) / KTILES;
    int m  = mt * 16 + (lane & 15);
    int kb = (lane < 16) ? 0 : 16;
    const float* hrow = h  + (size_t)m * D;
    const float* nrow = hN + (size_t)m * D;
    v16bf frag;
#pragma unroll
    for (int e = 0; e < 16; ++e) {
        int koff = kb + e;
        int kg = kt * 32 + koff;
        float v = (kg < D) ? hrow[kg] : nrow[kg - D];
        frag[e] = (__bf16)v;
    }
    *reinterpret_cast<v16bf*>(Ap + (size_t)t * 16) = frag;
}

// ---------------- WMMA GEMM: out[M,NCOLS] = A[M,1024] @ B + bias (opt ReLU) --
// block = 128 threads = 4 waves; each wave computes TPW adjacent 16x16 tiles,
// reusing one A fragment per k-step across TPW independent accumulator chains.
template <int NCOLS, int TPW, bool RELU>
__global__ __launch_bounds__(128)
void wmma_gemm(const __bf16* __restrict__ Ap, const __bf16* __restrict__ Bp,
               const float* __restrict__ bias, float* __restrict__ out) {
    const int lane = threadIdx.x & 31;
    const int wave = threadIdx.x >> 5;
    const int mt  = blockIdx.x;
    const int nt0 = (blockIdx.y * 4 + wave) * TPW;

    const __bf16* ap = Ap + ((size_t)mt  * KTILES * 32 + lane) * 16;
    const __bf16* bp = Bp + ((size_t)nt0 * KTILES * 32 + lane) * 16;
    const size_t bstride = (size_t)KTILES * 32 * 16;   // between adjacent N-tiles

    v8f c[TPW];
#pragma unroll
    for (int q = 0; q < TPW; ++q) c[q] = v8f{};

#pragma unroll 2
    for (int kt = 0; kt < KTILES; ++kt) {
        v16bf a = *reinterpret_cast<const v16bf*>(ap);
#pragma unroll
        for (int q = 0; q < TPW; ++q) {
            v16bf b = *reinterpret_cast<const v16bf*>(bp + q * bstride);
            c[q] = __builtin_amdgcn_wmma_f32_16x16x32_bf16(false, a, false, b,
                                                           (short)0, c[q], false, false);
        }
        ap += 32 * 16;
        bp += 32 * 16;
    }

    // C/D layout: lane L holds n = L%16; VGPR j holds m = j (+8 for upper half)
    const int mbase = mt * 16 + ((lane < 16) ? 0 : 8);
#pragma unroll
    for (int q = 0; q < TPW; ++q) {
        const int n = (nt0 + q) * 16 + (lane & 15);
        const float bv = bias[n];
#pragma unroll
        for (int j = 0; j < 8; ++j) {
            float v = c[q][j] + bv;
            if (RELU) v = fmaxf(v, 0.0f);
            out[(size_t)(mbase + j) * NCOLS + n] = v;
        }
    }
}

// ---------------------------------------------------------------------------
extern "C" void kernel_launch(void* const* d_in, const int* in_sizes, int n_in,
                              void* d_out, int out_size, void* d_ws, size_t ws_size,
                              hipStream_t stream) {
    const float* h   = (const float*)d_in[0];
    const float* W1  = (const float*)d_in[1];
    const float* b1  = (const float*)d_in[2];
    const float* W2  = (const float*)d_in[3];
    const float* b2  = (const float*)d_in[4];
    const int*   src = (const int*)d_in[5];
    const int*   dst = (const int*)d_in[6];
    float* out = (float*)d_out;

    char* ws = (char*)d_ws;
    size_t off = 0;
    auto wsalloc = [&](size_t bytes) -> void* {
        void* p = ws + off;
        off += (bytes + 255) & ~(size_t)255;
        return p;
    };
    int*    cnt       = (int*)wsalloc((size_t)N_NODES * sizeof(int));
    int*    row_start = (int*)wsalloc((size_t)N_NODES * sizeof(int));
    int*    cursor    = (int*)wsalloc((size_t)N_NODES * sizeof(int));
    int*    edge_src  = (int*)wsalloc((size_t)N_EDGES * sizeof(int));
    float*  hN        = (float*)wsalloc((size_t)N_NODES * D * sizeof(float));
    float*  h1        = (float*)wsalloc((size_t)N_NODES * D * sizeof(float));
    __bf16* Ap        = (__bf16*)wsalloc((size_t)N_NODES * KTOT * sizeof(__bf16));
    __bf16* B1p       = (__bf16*)wsalloc((size_t)KTOT * D * sizeof(__bf16));
    __bf16* B2p       = (__bf16*)wsalloc((size_t)KTOT * NCLS * sizeof(__bf16));

    const int packA_threads  = MTILES * KTILES * 32;        // 640,000
    const int packB1_threads = (D / 16) * KTILES * 32;      // 32,768
    const int packB2_threads = (NCLS / 16) * KTILES * 32;   // 8,192

    // ---- weight packing (independent of graph work) ----
    pack_B<<<(packB1_threads + 255) / 256, 256, 0, stream>>>(W1, B1p, D);
    pack_B<<<(packB2_threads + 255) / 256, 256, 0, stream>>>(W2, B2p, NCLS);

    // ---- CSR build (once; reused by both layers) ----
    zero_b32<<<64, 256, 0, stream>>>((unsigned*)cnt, N_NODES);
    degree_kernel<<<(N_EDGES + 255) / 256, 256, 0, stream>>>(dst, cnt);
    scan_kernel<<<1, 1024, 0, stream>>>(cnt, row_start, cursor);
    bucket_kernel<<<(N_EDGES + 255) / 256, 256, 0, stream>>>(src, dst, cursor, edge_src);

    // ---- layer 1 ----
    gather_mean<<<N_NODES, 256, 0, stream>>>(h, edge_src, row_start, cnt, hN);
    pack_A<<<(packA_threads + 255) / 256, 256, 0, stream>>>(h, hN, Ap);
    wmma_gemm<D, 4, true><<<dim3(MTILES, D / 256), 128, 0, stream>>>(Ap, B1p, b1, h1);

    // ---- layer 2 ----
    gather_mean<<<N_NODES, 256, 0, stream>>>(h1, edge_src, row_start, cnt, hN);
    pack_A<<<(packA_threads + 255) / 256, 256, 0, stream>>>(h1, hN, Ap);
    wmma_gemm<NCLS, 2, false><<<dim3(MTILES, 1), 128, 0, stream>>>(Ap, B2p, b2, out);
}
